// ViViTAE_23424751632833
// MI455X (gfx1250) — matrix operand order repaired and checked by reading, not compiled, not run on hardware
//
#include <hip/hip_runtime.h>
#include <hip/hip_bf16.h>

// ---------------------------------------------------------------------------
// Types / helpers
// ---------------------------------------------------------------------------
typedef __bf16 v16bf __attribute__((ext_vector_type(16)));
typedef float  v8f   __attribute__((ext_vector_type(8)));

__device__ __forceinline__ unsigned short f2bf(float f) {
  unsigned u = __builtin_bit_cast(unsigned, f);
  u += 0x7FFFu + ((u >> 16) & 1u);              // round-to-nearest-even
  return (unsigned short)(u >> 16);
}
__device__ __forceinline__ unsigned pk2(float a, float b) {
  return (unsigned)f2bf(a) | ((unsigned)f2bf(b) << 16);
}

// ---------------------------------------------------------------------------
// Generic WMMA GEMM:  C[M,N] = A[M,K] * op(B) (+bias[n]) (+aux[(m%auxMod),n]) (ReLU?)
//   A: f32 (converted to bf16 on LDS stage) or bf16
//   B: always bf16.  TB=true : B is W[N,K] row-major (X @ W^T).
//                    TB=false: B is B[K,N] row-major (transposed while staging).
//   Block tile 64x64, 4 waves (128 thr), each wave 32x32 = 4 wmma tiles, K-step 32.
//   Uniform fast path: 128-bit global loads + ds_store_b128, no per-lane guards.
// ---------------------------------------------------------------------------
#define LDSS 40   // LDS row stride in bf16 elems (80B: 16B-aligned rows, conflict-spread)

template<bool ABF, bool TB, bool OBF, bool RELU>
__global__ __launch_bounds__(128) void gemm_wmma_kernel(
    const void* __restrict__ Av, long lda, long sA,
    const unsigned short* __restrict__ B, long ldb, long sB,
    void* __restrict__ Cv, long ldc, long sC,
    const float* __restrict__ bias,
    const float* __restrict__ aux, int auxMod, long auxLd,
    int M, int N, int K)
{
  __shared__ unsigned short Al[64 * LDSS];
  __shared__ unsigned short Bl[64 * LDSS];

  const int tid  = threadIdx.x;
  const int wave = tid >> 5, lane = tid & 31;
  const int h    = lane >> 4, mr = lane & 15;
  const int wm   = wave >> 1, wn = wave & 1;
  const int bm   = blockIdx.y * 64, bn = blockIdx.x * 64;
  const long z   = blockIdx.z;
  const bool fullM = (bm + 64 <= M);
  const bool fullN = (bn + 64 <= N);

  const float*          Af = ABF ? nullptr : ((const float*)Av + z * sA);
  const unsigned short* Ab = ABF ? ((const unsigned short*)Av + z * sA) : nullptr;
  B += z * sB;
  float*          Cf = OBF ? nullptr : ((float*)Cv + z * sC);
  unsigned short* Cb = OBF ? ((unsigned short*)Cv + z * sC) : nullptr;

  // staging coordinates
  const int rm  = tid >> 1, ka0 = (tid & 1) * 16;   // A / B(W-layout): row + 16-elem half
  const int ncol = tid & 63, kb0 = (tid >> 6) * 16; // B[K,N] transpose loader

  v8f acc[2][2];
#pragma unroll
  for (int i = 0; i < 2; ++i)
#pragma unroll
    for (int j = 0; j < 2; ++j)
#pragma unroll
      for (int d = 0; d < 8; ++d) acc[i][j][d] = 0.0f;

  for (int k0 = 0; k0 < K; k0 += 32) {
    const bool kfull = (k0 + 32 <= K);

    // ---- stage A tile (64 x 32) -> LDS bf16
    {
      const int grow = bm + rm;
      if (fullM && kfull) {
        if (ABF) {
          const uint4* s = (const uint4*)(Ab + (long)grow * lda + k0 + ka0);
          uint4 a0 = s[0], a1 = s[1];
          *(uint4*)&Al[rm * LDSS + ka0]     = a0;
          *(uint4*)&Al[rm * LDSS + ka0 + 8] = a1;
        } else {
          const float4* s = (const float4*)(Af + (long)grow * lda + k0 + ka0);
          if (k0 + 32 < K) __builtin_prefetch(s + 8, 0, 0);
          float4 f0 = s[0], f1 = s[1], f2 = s[2], f3 = s[3];
          uint4 lo = {pk2(f0.x, f0.y), pk2(f0.z, f0.w), pk2(f1.x, f1.y), pk2(f1.z, f1.w)};
          uint4 hi = {pk2(f2.x, f2.y), pk2(f2.z, f2.w), pk2(f3.x, f3.y), pk2(f3.z, f3.w)};
          *(uint4*)&Al[rm * LDSS + ka0]     = lo;
          *(uint4*)&Al[rm * LDSS + ka0 + 8] = hi;
        }
      } else {
#pragma unroll 1
        for (int kk = 0; kk < 16; ++kk) {
          const int gk = k0 + ka0 + kk;
          unsigned short v = 0;
          if (grow < M && gk < K)
            v = ABF ? Ab[(long)grow * lda + gk] : f2bf(Af[(long)grow * lda + gk]);
          Al[rm * LDSS + ka0 + kk] = v;
        }
      }
    }
    // ---- stage B tile -> LDS as [n][k]
    if (TB) {
      const int gn = bn + rm;
      if (fullN && kfull) {
        const uint4* s = (const uint4*)(B + (long)gn * ldb + k0 + ka0);
        uint4 b0 = s[0], b1 = s[1];
        *(uint4*)&Bl[rm * LDSS + ka0]     = b0;
        *(uint4*)&Bl[rm * LDSS + ka0 + 8] = b1;
      } else {
#pragma unroll 1
        for (int kk = 0; kk < 16; ++kk) {
          const int gk = k0 + ka0 + kk;
          Bl[rm * LDSS + ka0 + kk] =
              (gn < N && gk < K) ? B[(long)gn * ldb + gk] : (unsigned short)0;
        }
      }
    } else {
      const int gn = bn + ncol;
      if (fullN && kfull) {
#pragma unroll
        for (int kk = 0; kk < 16; ++kk)
          Bl[ncol * LDSS + kb0 + kk] = B[(long)(k0 + kb0 + kk) * ldb + gn];
      } else {
#pragma unroll 1
        for (int kk = 0; kk < 16; ++kk) {
          const int gk = k0 + kb0 + kk;
          Bl[ncol * LDSS + kb0 + kk] =
              (gn < N && gk < K) ? B[(long)gk * ldb + gn] : (unsigned short)0;
        }
      }
    }
    __syncthreads();

    // ---- build fragments per ISA VGPR layouts and issue WMMA
    union FragU { unsigned u[8]; v16bf v; } fa[2], fb[2];
#pragma unroll
    for (int t2 = 0; t2 < 2; ++t2) {
      const int arow = (wm * 32 + t2 * 16 + mr) * LDSS;
      const int brow = (wn * 32 + t2 * 16 + mr) * LDSS;
#pragma unroll
      for (int p = 0; p < 8; ++p) {
        const int ka = 2 * (p & 3) + ((p >= 4) ? 16 : 0) + h * 8; // A 16x32 bf16 layout
        const int kb = 2 * (p & 3) + ((p >= 4) ? 16 : 0) + h * 8; // B 32x16 bf16 layout (same k order as A)
        fa[t2].u[p] = *(const unsigned*)&Al[arow + ka];
        fb[t2].u[p] = *(const unsigned*)&Bl[brow + kb];
      }
    }
#pragma unroll
    for (int i = 0; i < 2; ++i)
#pragma unroll
      for (int j = 0; j < 2; ++j)
        acc[i][j] = __builtin_amdgcn_wmma_f32_16x16x32_bf16(
            false, fa[i].v, false, fb[j].v, (short)0, acc[i][j], false, false);
    __syncthreads();
  }

  // ---- epilogue (C 16x16 f32 layout: vgpr d -> row h*8+d, col mr)
  if (fullM && fullN) {
#pragma unroll
    for (int i = 0; i < 2; ++i)
#pragma unroll
      for (int j = 0; j < 2; ++j) {
        const int col = bn + wn * 32 + j * 16 + mr;
#pragma unroll
        for (int d = 0; d < 8; ++d) {
          const int row = bm + wm * 32 + i * 16 + h * 8 + d;
          float v = acc[i][j][d];
          if (bias) v += bias[col];
          if (aux)  v += aux[(long)(row % auxMod) * auxLd + col];
          if (RELU) v = fmaxf(v, 0.0f);
          if (OBF) Cb[(long)row * ldc + col] = f2bf(v);
          else     Cf[(long)row * ldc + col] = v;
        }
      }
  } else {
#pragma unroll 1
    for (int i = 0; i < 2; ++i)
#pragma unroll 1
      for (int j = 0; j < 2; ++j) {
        const int col = bn + wn * 32 + j * 16 + mr;
#pragma unroll 1
        for (int d = 0; d < 8; ++d) {
          const int row = bm + wm * 32 + i * 16 + h * 8 + d;
          if (row < M && col < N) {
            float v = acc[i][j][d];
            if (bias) v += bias[col];
            if (aux)  v += aux[(long)(row % auxMod) * auxLd + col];
            if (RELU) v = fmaxf(v, 0.0f);
            if (OBF) Cb[(long)row * ldc + col] = f2bf(v);
            else     Cf[(long)row * ldc + col] = v;
          }
        }
      }
  }
}

// ---------------------------------------------------------------------------
// Elementwise / reduction kernels
// ---------------------------------------------------------------------------
__global__ void cast_f2bf_kernel(const float* __restrict__ in,
                                 unsigned short* __restrict__ out, long n) {
  long i = (long)blockIdx.x * blockDim.x + threadIdx.x;
  if (i < n) out[i] = f2bf(in[i]);
}

// im2col for Conv2d(k=s=16): x[8,16,3,384,384] -> Ap[73728,768] bf16
__global__ void im2col_kernel(const float* __restrict__ x,
                              unsigned short* __restrict__ ap, long total) {
  long i = (long)blockIdx.x * blockDim.x + threadIdx.x;
  if (i >= total) return;
  const int  k  = (int)(i % 768);
  const long r  = i / 768;
  const int  n  = (int)(r % 576);
  const long ft = r / 576;
  const int  t  = (int)(ft % 16);
  const int  b  = (int)(ft / 16);
  const int ii = n / 24, jj = n % 24;
  const int c = k >> 8, p = (k >> 4) & 15, q = k & 15;
  const long src = ((((long)b * 16 + t) * 3 + c) * 384 + (ii * 16 + p)) * 384 + (jj * 16 + q);
  ap[i] = f2bf(x[src]);
}

// per-row: D = softmax(S1*scale) - lam * softmax(S2*scale)   (bf16 out)
__global__ __launch_bounds__(64) void softmax_diff_kernel(
    const float* __restrict__ S1, const float* __restrict__ S2,
    const float* __restrict__ lamp, unsigned short* __restrict__ D,
    int Nk, float scale)
{
  __shared__ float red[64];
  const long row = blockIdx.x;
  const float* s1 = S1 + row * Nk;
  const float* s2 = S2 + row * Nk;
  unsigned short* dq = D + row * Nk;
  const int tid = threadIdx.x;

  float m1 = -3e38f, m2 = -3e38f;
  for (int j = tid; j < Nk; j += 64) {
    m1 = fmaxf(m1, s1[j] * scale);
    m2 = fmaxf(m2, s2[j] * scale);
  }
  red[tid] = m1; __syncthreads();
  for (int o = 32; o > 0; o >>= 1) { if (tid < o) red[tid] = fmaxf(red[tid], red[tid + o]); __syncthreads(); }
  m1 = red[0]; __syncthreads();
  red[tid] = m2; __syncthreads();
  for (int o = 32; o > 0; o >>= 1) { if (tid < o) red[tid] = fmaxf(red[tid], red[tid + o]); __syncthreads(); }
  m2 = red[0]; __syncthreads();

  float t1 = 0.f, t2 = 0.f;
  for (int j = tid; j < Nk; j += 64) {
    t1 += __expf(s1[j] * scale - m1);
    t2 += __expf(s2[j] * scale - m2);
  }
  red[tid] = t1; __syncthreads();
  for (int o = 32; o > 0; o >>= 1) { if (tid < o) red[tid] += red[tid + o]; __syncthreads(); }
  t1 = red[0]; __syncthreads();
  red[tid] = t2; __syncthreads();
  for (int o = 32; o > 0; o >>= 1) { if (tid < o) red[tid] += red[tid + o]; __syncthreads(); }
  t2 = red[0]; __syncthreads();

  const float lam = *lamp;
  const float r1 = 1.0f / t1, r2 = lam / t2;
  for (int j = tid; j < Nk; j += 64)
    dq[j] = f2bf(__expf(s1[j] * scale - m1) * r1 - __expf(s2[j] * scale - m2) * r2);
}

__global__ __launch_bounds__(128) void layernorm_kernel(
    const float* __restrict__ X, const float* __restrict__ w,
    const float* __restrict__ b, float* __restrict__ Y, int E)
{
  __shared__ float red[128];
  const long row = blockIdx.x;
  const float* x = X + row * E;
  float* y = Y + row * E;
  const int tid = threadIdx.x;
  float s = 0.f, ss = 0.f;
  for (int j = tid; j < E; j += 128) { float v = x[j]; s += v; ss += v * v; }
  red[tid] = s; __syncthreads();
  for (int o = 64; o > 0; o >>= 1) { if (tid < o) red[tid] += red[tid + o]; __syncthreads(); }
  s = red[0]; __syncthreads();
  red[tid] = ss; __syncthreads();
  for (int o = 64; o > 0; o >>= 1) { if (tid < o) red[tid] += red[tid + o]; __syncthreads(); }
  ss = red[0]; __syncthreads();
  const float mu = s / E;
  const float rstd = rsqrtf(ss / E - mu * mu + 1e-5f);
  for (int j = tid; j < E; j += 128) y[j] = (x[j] - mu) * rstd * w[j] + b[j];
}

// frames[f,e] = mean_n X[f*rpf+n, e] + temb[f % tmod, e]
__global__ __launch_bounds__(64) void mean_kernel(
    const float* __restrict__ X, const float* __restrict__ temb,
    float* __restrict__ out, int rpf, int tmod)
{
  const int e = blockIdx.x * 64 + threadIdx.x;
  const int f = blockIdx.y;
  float s = 0.f;
  for (int n = 0; n < rpf; ++n) s += X[((long)f * rpf + n) * 512 + e];
  out[(long)f * 512 + e] = s / (float)rpf + temb[(long)(f % tmod) * 512 + e];
}

// torch transpose(0,1).view permutation
__global__ __launch_bounds__(128) void scramble_kernel(
    const float* __restrict__ in, float* __restrict__ out, int Bc, int Tn, int Nn)
{
  const int Rp = blockIdx.x;
  const int bp = Rp / (Tn * Nn);
  const int t  = (Rp / Nn) % Tn;
  const int np = Rp % Nn;
  const int L  = bp * Nn + np;
  const int n  = L / Bc, b = L % Bc;
  const long src = ((long)(b * Tn + t) * Nn + n) * 512;
  const long dst = (long)Rp * 512;
  for (int c = threadIdx.x; c < 512; c += 128) out[dst + c] = in[src + c];
}

// ---------------------------------------------------------------------------
// Host-side launch helpers
// ---------------------------------------------------------------------------
template<bool ABF, bool TB, bool OBF, bool RELU>
static void gemm(hipStream_t st,
                 const void* A, long lda, long sA,
                 const void* B, long ldb, long sB,
                 void* C, long ldc, long sC,
                 const float* bias, const float* aux, int auxMod, long auxLd,
                 int M, int N, int K, int batch)
{
  dim3 grid((unsigned)((N + 63) / 64), (unsigned)((M + 63) / 64), (unsigned)batch);
  dim3 block(128);
  hipLaunchKernelGGL(HIP_KERNEL_NAME(gemm_wmma_kernel<ABF, TB, OBF, RELU>),
                     grid, block, 0, st,
                     A, lda, sA, (const unsigned short*)B, ldb, sB,
                     C, ldc, sC, bias, aux, auxMod, auxLd, M, N, K);
}

static void castbf(hipStream_t st, const float* in, unsigned short* out, long n) {
  hipLaunchKernelGGL(cast_f2bf_kernel, dim3((unsigned)((n + 255) / 256)), dim3(256), 0, st, in, out, n);
}

extern "C" void kernel_launch(void* const* d_in, const int* in_sizes, int n_in,
                              void* d_out, int out_size, void* d_ws, size_t ws_size,
                              hipStream_t stream)
{
  (void)in_sizes; (void)n_in; (void)out_size; (void)ws_size;
  // ---- inputs (setup_inputs dict order, dicts flattened in insertion order)
  const float* x      = (const float*)d_in[0];
  const float* conv_w = (const float*)d_in[1];
  const float* conv_b = (const float*)d_in[2];
  const float* pos    = (const float*)d_in[3];
  const float* temb   = (const float*)d_in[4];
  const float* sWqkv = (const float*)d_in[5];
  const float* sbqkv = (const float*)d_in[6];
  const float* sWo   = (const float*)d_in[7];
  const float* sbo   = (const float*)d_in[8];
  const float* slam  = (const float*)d_in[9];
  const float* sn1w  = (const float*)d_in[10];
  const float* sn1b  = (const float*)d_in[11];
  const float* sW1   = (const float*)d_in[12];
  const float* sb1   = (const float*)d_in[13];
  const float* sW2   = (const float*)d_in[14];
  const float* sb2   = (const float*)d_in[15];
  const float* sn2w  = (const float*)d_in[16];
  const float* sn2b  = (const float*)d_in[17];
  const float* tWqkv = (const float*)d_in[18];
  const float* tbqkv = (const float*)d_in[19];
  const float* tWo   = (const float*)d_in[20];
  const float* tbo   = (const float*)d_in[21];
  const float* tlam  = (const float*)d_in[22];
  const float* tn1w  = (const float*)d_in[23];
  const float* tn1b  = (const float*)d_in[24];
  const float* tW1   = (const float*)d_in[25];
  const float* tb1   = (const float*)d_in[26];
  const float* tW2   = (const float*)d_in[27];
  const float* tb2   = (const float*)d_in[28];
  const float* tn2w  = (const float*)d_in[29];
  const float* tn2b  = (const float*)d_in[30];
  const float* Wout  = (const float*)d_in[31];
  const float* bout  = (const float*)d_in[32];
  float* out = (float*)d_out;

  // ---- dims
  const int Msp = 73728;         // b*t*N = 8*16*576 tokens
  const int E = 512, Ns = 576, Fr = 128, QN = 1536, Kp = 768;
  const int Mt = 128, Nt = 16;   // temporal: rows = b*t, seq len 16
  const float scale = 0.125f;    // 1/sqrt(64)

  // ---- workspace layout (lifetime-aliased regions)
  char* ws = (char*)d_ws;
  const size_t oR1 = 0;                       // Ap | S1 | AO | Y1 | Y2
  const size_t oR2 = oR1 + 169869312;         // S2 | XN | XS
  const size_t oR3 = oR2 + 169869312;         // QKV | AOp | Hb
  const size_t oR4 = oR3 + 226492416;         // X (encoder input, residual)
  const size_t oR5 = oR4 + 150994944;         // DM (bf16 diff-softmax)
  size_t o = oR5 + 84934656;
  auto bump = [&](size_t bytes) { size_t r = o; o += (bytes + 255) & ~(size_t)255; return r; };

  unsigned short* WcB   = (unsigned short*)(ws + bump(393216 * 2));
  unsigned short* sWqkvB= (unsigned short*)(ws + bump(786432 * 2));
  unsigned short* sWoB  = (unsigned short*)(ws + bump(262144 * 2));
  unsigned short* sW1B  = (unsigned short*)(ws + bump(262144 * 2));
  unsigned short* sW2B  = (unsigned short*)(ws + bump(262144 * 2));
  unsigned short* tWqkvB= (unsigned short*)(ws + bump(786432 * 2));
  unsigned short* tWoB  = (unsigned short*)(ws + bump(262144 * 2));
  unsigned short* tW1B  = (unsigned short*)(ws + bump(262144 * 2));
  unsigned short* tW2B  = (unsigned short*)(ws + bump(262144 * 2));
  unsigned short* WoutB = (unsigned short*)(ws + bump(262144 * 2));
  float* T0   = (float*)(ws + bump(Mt * E * 4));
  unsigned short* QKVt = (unsigned short*)(ws + bump((size_t)Mt * QN * 2));
  float* S1t  = (float*)(ws + bump(8 * 16 * 16 * 4));
  float* S2t  = (float*)(ws + bump(8 * 16 * 16 * 4));
  unsigned short* DMt = (unsigned short*)(ws + bump(8 * 16 * 16 * 2));
  float* AOt  = (float*)(ws + bump(Mt * E * 4));
  float* AOpt = (float*)(ws + bump(Mt * E * 4));
  float* Y1t  = (float*)(ws + bump(Mt * E * 4));
  float* XNt  = (float*)(ws + bump(Mt * E * 4));
  unsigned short* Htb = (unsigned short*)(ws + bump(Mt * E * 2));
  float* Y2t  = (float*)(ws + bump(Mt * E * 4));
  float* XTo  = (float*)(ws + bump(Mt * E * 4));

  unsigned short* Ap  = (unsigned short*)(ws + oR1);
  float*          S1  = (float*)(ws + oR1);
  float*          AO  = (float*)(ws + oR1);
  float*          Y1  = (float*)(ws + oR1);
  float*          Y2  = (float*)(ws + oR1);
  float*          S2  = (float*)(ws + oR2);
  float*          XN  = (float*)(ws + oR2);
  float*          XS  = (float*)(ws + oR2);
  unsigned short* QKV = (unsigned short*)(ws + oR3);
  float*          AOp = (float*)(ws + oR3);
  unsigned short* Hb  = (unsigned short*)(ws + oR3);
  float*          X   = (float*)(ws + oR4);
  unsigned short* DM  = (unsigned short*)(ws + oR5);

  // ---- weight casts to bf16
  castbf(stream, conv_w, WcB, 393216);
  castbf(stream, sWqkv, sWqkvB, 786432);
  castbf(stream, sWo, sWoB, 262144);
  castbf(stream, sW1, sW1B, 262144);
  castbf(stream, sW2, sW2B, 262144);
  castbf(stream, tWqkv, tWqkvB, 786432);
  castbf(stream, tWo, tWoB, 262144);
  castbf(stream, tW1, tW1B, 262144);
  castbf(stream, tW2, tW2B, 262144);
  castbf(stream, Wout, WoutB, 262144);

  // ---- patch embed: im2col + GEMM (+conv_b +pos)
  {
    long tot = (long)Msp * Kp;
    hipLaunchKernelGGL(im2col_kernel, dim3((unsigned)((tot + 255) / 256)), dim3(256), 0, stream, x, Ap, tot);
  }
  gemm<true, true, false, false>(stream, Ap, Kp, 0, WcB, Kp, 0, X, E, 0,
                                 conv_b, pos, Ns, E, Msp, E, Kp, 1);

  // ================= spatial encoder (128 independent 576-token frames) ====
  gemm<false, true, true, false>(stream, X, E, 0, sWqkvB, E, 0, QKV, QN, 0,
                                 sbqkv, nullptr, 1, 0, Msp, QN, E, 1);
  gemm<true, true, false, false>(stream, QKV + 0,   QN, (long)Ns * QN, QKV + 512, QN, (long)Ns * QN,
                                 S1, Ns, (long)Ns * Ns, nullptr, nullptr, 1, 0, Ns, Ns, 256, Fr);
  gemm<true, true, false, false>(stream, QKV + 256, QN, (long)Ns * QN, QKV + 768, QN, (long)Ns * QN,
                                 S2, Ns, (long)Ns * Ns, nullptr, nullptr, 1, 0, Ns, Ns, 256, Fr);
  hipLaunchKernelGGL(softmax_diff_kernel, dim3(Fr * Ns), dim3(64), 0, stream,
                     S1, S2, slam, DM, Ns, scale);
  gemm<true, false, false, false>(stream, DM, Ns, (long)Ns * Ns, QKV + 1024, QN, (long)Ns * QN,
                                  AO, E, (long)Ns * E, nullptr, nullptr, 1, 0, Ns, E, Ns, Fr);
  hipLaunchKernelGGL(scramble_kernel, dim3(Msp), dim3(128), 0, stream, AO, AOp, 8, 16, Ns);
  gemm<false, true, false, false>(stream, AOp, E, 0, sWoB, E, 0, Y1, E, 0,
                                  sbo, X, Msp, E, Msp, E, E, 1);
  hipLaunchKernelGGL(layernorm_kernel, dim3(Msp), dim3(128), 0, stream, Y1, sn1w, sn1b, XN, E);
  gemm<false, true, true, true>(stream, XN, E, 0, sW1B, E, 0, Hb, E, 0,
                                sb1, nullptr, 1, 0, Msp, E, E, 1);
  gemm<true, true, false, false>(stream, Hb, E, 0, sW2B, E, 0, Y2, E, 0,
                                 sb2, XN, Msp, E, Msp, E, E, 1);
  hipLaunchKernelGGL(layernorm_kernel, dim3(Msp), dim3(128), 0, stream, Y2, sn2w, sn2b, XS, E);

  // ---- mean over tokens + temb -> temporal input T0 [128,512]
  hipLaunchKernelGGL(mean_kernel, dim3(8, Fr), dim3(64), 0, stream, XS, temb, T0, Ns, 16);

  // ================= temporal encoder (8 batches of 16 tokens) =============
  gemm<false, true, true, false>(stream, T0, E, 0, tWqkvB, E, 0, QKVt, QN, 0,
                                 tbqkv, nullptr, 1, 0, Mt, QN, E, 1);
  gemm<true, true, false, false>(stream, QKVt + 0,   QN, (long)Nt * QN, QKVt + 512, QN, (long)Nt * QN,
                                 S1t, Nt, (long)Nt * Nt, nullptr, nullptr, 1, 0, Nt, Nt, 256, 8);
  gemm<true, true, false, false>(stream, QKVt + 256, QN, (long)Nt * QN, QKVt + 768, QN, (long)Nt * QN,
                                 S2t, Nt, (long)Nt * Nt, nullptr, nullptr, 1, 0, Nt, Nt, 256, 8);
  hipLaunchKernelGGL(softmax_diff_kernel, dim3(8 * Nt), dim3(64), 0, stream,
                     S1t, S2t, tlam, DMt, Nt, scale);
  gemm<true, false, false, false>(stream, DMt, Nt, (long)Nt * Nt, QKVt + 1024, QN, (long)Nt * QN,
                                  AOt, E, (long)Nt * E, nullptr, nullptr, 1, 0, Nt, E, Nt, 8);
  hipLaunchKernelGGL(scramble_kernel, dim3(Mt), dim3(128), 0, stream, AOt, AOpt, 8, 1, Nt);
  gemm<false, true, false, false>(stream, AOpt, E, 0, tWoB, E, 0, Y1t, E, 0,
                                  tbo, T0, Mt, E, Mt, E, E, 1);
  hipLaunchKernelGGL(layernorm_kernel, dim3(Mt), dim3(128), 0, stream, Y1t, tn1w, tn1b, XNt, E);
  gemm<false, true, true, true>(stream, XNt, E, 0, tW1B, E, 0, Htb, E, 0,
                                tb1, nullptr, 1, 0, Mt, E, E, 1);
  gemm<true, true, false, false>(stream, Htb, E, 0, tW2B, E, 0, Y2t, E, 0,
                                 tb2, XNt, Mt, E, Mt, E, E, 1);
  hipLaunchKernelGGL(layernorm_kernel, dim3(Mt), dim3(128), 0, stream, Y2t, tn2w, tn2b, XTo, E);

  // ---- final projection -> d_out [8,16,512]
  gemm<false, true, false, false>(stream, XTo, E, 0, WoutB, E, 0, out, E, 0,
                                  bout, nullptr, 1, 0, Mt, E, E, 1);
}
